// PerceiverResampler_22900765623147
// MI455X (gfx1250) — matrix operand order repaired and checked by reading, not compiled, not run on hardware
//
#include <hip/hip_runtime.h>
#include <math.h>

// ---------------- problem constants ----------------
constexpr int Bq     = 8;
constexpr int Nn     = 4096;
constexpr int Dd     = 1024;
constexpr int DEPTHq = 4;
constexpr int HEADSq = 16;
constexpr int DHq    = 64;
constexpr int INNERq = 1024;
constexpr int NLAT   = 68;      // 4 mean-pooled + 64 latents
constexpr int FFq    = 4096;
constexpr float EPSq = 1e-5f;
constexpr int NCH    = 66;      // ceil(4164/64): 64 x-chunks + 2 latent chunks

typedef __attribute__((ext_vector_type(16))) __bf16 v16bf;
typedef __attribute__((ext_vector_type(8)))  float  v8f;

union frag_u {            // WMMA 16-bit fragment viewed as 8 dwords
  v16bf v;
  unsigned u[8];
};

// K offset of element pair p (elements 2p,2p+1) for 16-bit 16x32 A matrix
// (contiguous order k = 16g + s, matching B); always even -> dword-aligned pair loads.
__device__ __forceinline__ int a_kpair(int p, int g) {
  int k = 2 * p;
  return k + 16 * g;
}

__device__ __forceinline__ float block_sum(float v, float* red) {
  int tid = threadIdx.x;
  red[tid] = v;
  __syncthreads();
  for (int s = 128; s > 0; s >>= 1) {
    if (tid < s) red[tid] += red[tid + s];
    __syncthreads();
  }
  float r = red[0];
  __syncthreads();
  return r;
}

// ---------------- elementwise / prep kernels ----------------
__global__ void cast_f32_to_bf16(const float* __restrict__ in, __bf16* __restrict__ out, long n) {
  long i = (long)blockIdx.x * 256 + threadIdx.x;
  if (i < n) out[i] = (__bf16)in[i];
}

// xp = x + pos_emb; xhat = (xp - mean)*rstd  (per-token, bf16 out). One block/token.
__global__ __launch_bounds__(256) void xhat_kernel(const float* __restrict__ x,
                                                   const float* __restrict__ pos,
                                                   __bf16* __restrict__ xhat) {
  __shared__ float red[256];
  long t = blockIdx.x;              // b*N + n
  long n = t % Nn;
  const float* xr = x + t * Dd;
  const float* pr = pos + n * Dd;
  float v[4], s = 0.f;
#pragma unroll
  for (int i = 0; i < 4; ++i) {
    int c = threadIdx.x + 256 * i;
    v[i] = xr[c] + pr[c];
    s += v[i];
  }
  float mean = block_sum(s, red) / Dd;
  float s2 = 0.f;
#pragma unroll
  for (int i = 0; i < 4; ++i) { float d = v[i] - mean; s2 += d * d; }
  float rstd = rsqrtf(block_sum(s2, red) / Dd + EPSq);
#pragma unroll
  for (int i = 0; i < 4; ++i) {
    int c = threadIdx.x + 256 * i;
    xhat[t * Dd + c] = (__bf16)((v[i] - mean) * rstd);
  }
}

// mp[b][d] = mean over tokens of x
__global__ void mp_mean(const float* __restrict__ x, float* __restrict__ mp) {
  int i = blockIdx.x * 256 + threadIdx.x;     // B*D
  if (i < Bq * Dd) {
    int b = i / Dd, d = i % Dd;
    const float* p = x + (long)b * Nn * Dd + d;
    float s = 0.f;
    for (int n = 0; n < Nn; ++n) s += p[(long)n * Dd];
    mp[i] = s / Nn;
  }
}

// generic row LN: optional exact-gelu pre-transform, optional gamma/beta,
// optional bf16 and/or f32 outputs. One block per row.
template <int EPT>
__global__ __launch_bounds__(256) void ln_rows(const float* __restrict__ in,
                                               const float* __restrict__ gamma,
                                               const float* __restrict__ beta,
                                               __bf16* __restrict__ outb,
                                               float* __restrict__ outf,
                                               int Dsz, int do_gelu) {
  __shared__ float red[256];
  long row = blockIdx.x;
  const float* src = in + row * (long)Dsz;
  float v[EPT], s = 0.f;
#pragma unroll
  for (int i = 0; i < EPT; ++i) {
    int c = threadIdx.x + 256 * i;
    float xv = src[c];
    if (do_gelu) xv = 0.5f * xv * (1.f + erff(xv * 0.70710678118654752f));
    v[i] = xv; s += xv;
  }
  float mean = block_sum(s, red) / Dsz;
  float s2 = 0.f;
#pragma unroll
  for (int i = 0; i < EPT; ++i) { float d = v[i] - mean; s2 += d * d; }
  float rstd = rsqrtf(block_sum(s2, red) / Dsz + EPSq);
#pragma unroll
  for (int i = 0; i < EPT; ++i) {
    int c = threadIdx.x + 256 * i;
    float y = (v[i] - mean) * rstd;
    if (gamma) y *= gamma[c];
    if (beta) y += beta[c];
    if (outb) outb[row * (long)Dsz + c] = (__bf16)y;
    if (outf) outf[row * (long)Dsz + c] = y;
  }
}

// lat += LN(proj, w, b)   (D=1024, one block per row)
__global__ __launch_bounds__(256) void postattn_ln_res(const float* __restrict__ proj,
                                                       const float* __restrict__ w,
                                                       const float* __restrict__ b,
                                                       float* __restrict__ lat) {
  __shared__ float red[256];
  long row = blockIdx.x;
  const float* src = proj + row * Dd;
  float v[4], s = 0.f;
#pragma unroll
  for (int i = 0; i < 4; ++i) {
    int c = threadIdx.x + 256 * i;
    v[i] = src[c]; s += v[i];
  }
  float mean = block_sum(s, red) / Dd;
  float s2 = 0.f;
#pragma unroll
  for (int i = 0; i < 4; ++i) { float d = v[i] - mean; s2 += d * d; }
  float rstd = rsqrtf(block_sum(s2, red) / Dd + EPSq);
#pragma unroll
  for (int i = 0; i < 4; ++i) {
    int c = threadIdx.x + 256 * i;
    lat[row * Dd + c] += (v[i] - mean) * rstd * w[c] + b[c];
  }
}

__global__ void add_inplace(float* __restrict__ dst, const float* __restrict__ src, long n) {
  long i = (long)blockIdx.x * 256 + threadIdx.x;
  if (i < n) dst[i] += src[i];
}

// lat init: rows 0..3 from mp_lat (B x 4*D), rows 4..67 from latents (64 x D)
__global__ void lat_init(const float* __restrict__ mplat, const float* __restrict__ latents,
                         float* __restrict__ lat) {
  long i = (long)blockIdx.x * 256 + threadIdx.x;
  if (i < (long)Bq * NLAT * Dd) {
    int d = (int)(i % Dd);
    int r = (int)((i / Dd) % NLAT);
    int b = (int)(i / ((long)NLAT * Dd));
    lat[i] = (r < 4) ? mplat[(long)b * 4 * Dd + r * Dd + d] : latents[(long)(r - 4) * Dd + d];
  }
}

// wkv_folded[d][c] = ln_x_w[d] * wkv[d][c]  (bf16 out)
__global__ void fold_wkv(const float* __restrict__ wkv, const float* __restrict__ lnw,
                         __bf16* __restrict__ out) {
  long i = (long)blockIdx.x * 256 + threadIdx.x;
  if (i < (long)Dd * 2 * INNERq) {
    int k = (int)(i / (2 * INNERq));
    out[i] = (__bf16)(wkv[i] * lnw[k]);
  }
}

// bias[c] = sum_d ln_x_b[d] * wkv[d][c]
__global__ void kv_bias(const float* __restrict__ wkv, const float* __restrict__ lnb,
                        float* __restrict__ bias) {
  int c = blockIdx.x * 256 + threadIdx.x;
  if (c < 2 * INNERq) {
    float s = 0.f;
    for (int k = 0; k < Dd; ++k) s += lnb[k] * wkv[(long)k * 2 * INNERq + c];
    bias[c] = s;
  }
}

// small fp32 GEMM: C(MxN) = A(MxK) @ W(KxN) + bias  (tiny mean-pool projection)
__global__ void small_gemm_f32(const float* __restrict__ A, const float* __restrict__ W,
                               const float* __restrict__ bias, float* __restrict__ C,
                               int M, int K, int N) {
  int c = blockIdx.x * 256 + threadIdx.x;
  int r = blockIdx.y;
  if (c < N && r < M) {
    float s = bias ? bias[c] : 0.f;
    const float* a = A + (long)r * K;
    for (int k = 0; k < K; ++k) s += a[k] * W[(long)k * N + c];
    C[(long)r * N + c] = s;
  }
}

// ---------------- WMMA bf16 GEMM: C = (A @ B) * scale + bias ----------------
// A: MxK bf16 row-major, B: KxN bf16 row-major. N%128==0, K%32==0, M guarded.
// A tile staged via global_load_async_to_lds_b128 (ASYNCcnt path);
// B tile staged transposed so B fragments are contiguous b128 LDS loads.
constexpr int GBM = 128, GBN = 128, GBK = 32;

__global__ __launch_bounds__(256) void gemm_bf16(const __bf16* __restrict__ A,
                                                 const __bf16* __restrict__ Bm,
                                                 void* __restrict__ C,
                                                 const float* __restrict__ bias,
                                                 int M, int N, int K,
                                                 float out_scale, int store_bf16) {
  __shared__ __align__(16) __bf16 As[GBM][GBK];       // row-major (M x K)
  __shared__ __align__(16) __bf16 Bs_t[GBN][GBK];     // transposed (N x K)
  int tid  = threadIdx.x;
  int lane = tid & 31, w = tid >> 5;
  int g = lane >> 4, col = lane & 15;
  int m0 = blockIdx.x * GBM;
  int n0 = blockIdx.y * GBN;

  v8f acc[8];
#pragma unroll
  for (int t = 0; t < 8; ++t)
#pragma unroll
    for (int r = 0; r < 8; ++r) acc[t][r] = 0.f;

  for (int kk = 0; kk < K; kk += GBK) {
    {   // stage A: 128 rows x 32 cols, 2 threads/row x 32B, async direct-to-LDS
      int r = tid >> 1;
      int off = (tid & 1) * 16;
      int row = m0 + r;
      __bf16* dst = &As[r][off];
      if (row < M) {
        const __bf16* src = A + (long)row * K + kk + off;
        unsigned la = (unsigned)(unsigned long long)(void*)dst;  // LDS = addr[31:0]
        asm volatile("global_load_async_to_lds_b128 %0, %1, off"
                     :: "v"(la), "v"(src) : "memory");
        asm volatile("global_load_async_to_lds_b128 %0, %1, off offset:16"
                     :: "v"(la), "v"(src) : "memory");
        if (kk + GBK < K) __builtin_prefetch(src + GBK, 0, 0);
      } else {
        float4 z = {0.f, 0.f, 0.f, 0.f};
        *(float4*)dst = z; *(float4*)(dst + 8) = z;
      }
    }
    {   // stage B transposed: read 16 contiguous cols of row kk+r, scatter to Bs_t
      int r = tid >> 3;          // k row 0..31
      int off = (tid & 7) * 16;  // n start
      const unsigned* sp = (const unsigned*)(Bm + (long)(kk + r) * N + n0 + off);
      unsigned tmp[8];
#pragma unroll
      for (int p = 0; p < 8; ++p) tmp[p] = sp[p];
      const __bf16* tb = (const __bf16*)tmp;
#pragma unroll
      for (int i = 0; i < 16; ++i) Bs_t[off + i][r] = tb[i];
    }
    asm volatile("s_wait_asynccnt 0x0" ::: "memory");
    __syncthreads();

    frag_u a;
#pragma unroll
    for (int p = 0; p < 8; ++p)
      a.u[p] = *(const unsigned*)&As[16 * w + col][a_kpair(p, g)];
#pragma unroll
    for (int tc = 0; tc < 8; ++tc) {
      frag_u b;
      const unsigned* bp = (const unsigned*)&Bs_t[16 * tc + col][16 * g];
#pragma unroll
      for (int p = 0; p < 8; ++p) b.u[p] = bp[p];
      acc[tc] = __builtin_amdgcn_wmma_f32_16x16x32_bf16(false, a.v, false, b.v,
                                                        (short)0, acc[tc], false, false);
    }
    __syncthreads();
  }

#pragma unroll
  for (int tc = 0; tc < 8; ++tc) {
    int cg = n0 + 16 * tc + col;
    float bv = bias ? bias[cg] : 0.f;
#pragma unroll
    for (int r = 0; r < 8; ++r) {
      int rg = m0 + 16 * w + r + 8 * g;
      if (rg < M) {
        float v = acc[tc][r] * out_scale + bv;
        if (store_bf16) ((__bf16*)C)[(long)rg * N + cg] = (__bf16)v;
        else            ((float*)C)[(long)rg * N + cg] = v;
      }
    }
  }
}

// ---------------- flash attention (per batch*head block) ----------------
// q: (B*68, 1024) bf16 pre-scaled. kvx: (B*4096, 2048) bf16. kvlat: (B*68, 2048) bf16.
// out: (B*68, 1024) bf16. grid = (HEADS, B), 256 threads.
__global__ __launch_bounds__(256) void attn_kernel(const __bf16* __restrict__ q,
                                                   const __bf16* __restrict__ kvx,
                                                   const __bf16* __restrict__ kvlat,
                                                   __bf16* __restrict__ out) {
  __shared__ __align__(16) float  s_s[80][64];   // S tile (fp32)
  __shared__ __align__(16) __bf16 p_s[80][64];   // probabilities (bf16)
  __shared__ __align__(16) __bf16 v_t[64][64];   // V chunk transposed (dh x key)
  __shared__ float alpha_s[80];
  __shared__ float l_s[80];

  int h  = blockIdx.x;
  int bb = blockIdx.y;
  int tid = threadIdx.x;
  int lane = tid & 31, w = tid >> 5;
  int g = lane >> 4, col = lane & 15;
  int tcw    = w & 3;
  int trbase = (w < 4) ? 0 : 3;
  int ntr    = (w < 4) ? 3 : 2;

  // Q A-fragments (rows padded 68->80), kept in registers
  frag_u a_q[5][2];
#pragma unroll
  for (int tr = 0; tr < 5; ++tr) {
    int qi = 16 * tr + col;
    const __bf16* qrow = q + ((long)bb * NLAT + qi) * INNERq + h * DHq;
#pragma unroll
    for (int kb = 0; kb < 2; ++kb)
#pragma unroll
      for (int p = 0; p < 8; ++p)
        a_q[tr][kb].u[p] = (qi < NLAT)
            ? *(const unsigned*)(qrow + 32 * kb + a_kpair(p, g)) : 0u;
  }

  v8f o_acc[3];
#pragma unroll
  for (int i = 0; i < 3; ++i)
#pragma unroll
    for (int r = 0; r < 8; ++r) o_acc[i][r] = 0.f;

  float m_run = -INFINITY, l_run = 0.f;

  for (int c = 0; c < NCH; ++c) {
    int j0 = c * 64;
    const __bf16* kbase;
    int valid;
    if (j0 < Nn) { kbase = kvx + ((long)bb * Nn + j0) * 2048 + h * DHq; valid = 64; }
    else {
      int jl = j0 - Nn;
      kbase = kvlat + ((long)bb * NLAT + jl) * 2048 + h * DHq;
      valid = NLAT - jl; if (valid > 64) valid = 64;
    }
    const __bf16* vbase = kbase + INNERq;

    // ---- S = Q K^T for this wave's tile-col ----
    v8f sacc[3];
#pragma unroll
    for (int i = 0; i < 3; ++i)
#pragma unroll
      for (int r = 0; r < 8; ++r) sacc[i][r] = 0.f;

    int local = 16 * tcw + col;
    bool okr = local < valid;
    const __bf16* krow = kbase + (long)local * 2048;
#pragma unroll
    for (int kb = 0; kb < 2; ++kb) {
      frag_u bk;
      const unsigned* kp = (const unsigned*)(krow + 32 * kb + 16 * g);
#pragma unroll
      for (int p = 0; p < 8; ++p) bk.u[p] = okr ? kp[p] : 0u;
      for (int i = 0; i < ntr; ++i)
        sacc[i] = __builtin_amdgcn_wmma_f32_16x16x32_bf16(false, a_q[trbase + i][kb].v,
                                                          false, bk.v, (short)0,
                                                          sacc[i], false, false);
    }
    for (int i = 0; i < ntr; ++i) {
      int tr = trbase + i;
#pragma unroll
      for (int r = 0; r < 8; ++r) s_s[16 * tr + r + 8 * g][16 * tcw + col] = sacc[i][r];
    }

    // ---- stage V chunk transposed (dh x key) ----
    {
      int r = tid >> 2;              // key 0..63
      int off = (tid & 3) * 16;      // dh start
      if (r < valid) {
        const unsigned* sp = (const unsigned*)(vbase + (long)r * 2048 + off);
        unsigned tmp[8];
#pragma unroll
        for (int p = 0; p < 8; ++p) tmp[p] = sp[p];
        const __bf16* tb = (const __bf16*)tmp;
#pragma unroll
        for (int i = 0; i < 16; ++i) v_t[off + i][r] = tb[i];
      } else {
#pragma unroll
        for (int i = 0; i < 16; ++i) v_t[off + i][r] = (__bf16)0.f;
      }
    }
    __syncthreads();

    // ---- online softmax: thread tid owns row tid (tid < 80) ----
    if (tid < 80) {
      float cm = -INFINITY;
      for (int cc = 0; cc < valid; ++cc) cm = fmaxf(cm, s_s[tid][cc]);
      float mnew = fmaxf(m_run, cm);
      float alpha = __expf(m_run - mnew);      // 0 on first chunk (m_run=-inf)
      float ps = 0.f;
      for (int cc = 0; cc < valid; ++cc) {
        float p = __expf(s_s[tid][cc] - mnew);
        ps += p;
        p_s[tid][cc] = (__bf16)p;
      }
      for (int cc = valid; cc < 64; ++cc) p_s[tid][cc] = (__bf16)0.f;
      l_run = l_run * alpha + ps;
      m_run = mnew;
      alpha_s[tid] = alpha;
      l_s[tid] = l_run;
    }
    __syncthreads();

    // ---- O = O*alpha + P @ V (wave: dh tile-col tcw, rows trbase..) ----
    for (int i = 0; i < ntr; ++i) {
      int tr = trbase + i;
#pragma unroll
      for (int r = 0; r < 8; ++r) o_acc[i][r] *= alpha_s[16 * tr + r + 8 * g];
    }
#pragma unroll
    for (int kb2 = 0; kb2 < 2; ++kb2) {
      frag_u bv;
      const unsigned* vp = (const unsigned*)&v_t[16 * tcw + col][32 * kb2 + 16 * g];
#pragma unroll
      for (int p = 0; p < 8; ++p) bv.u[p] = vp[p];
      for (int i = 0; i < ntr; ++i) {
        int tr = trbase + i;
        frag_u ap;
#pragma unroll
        for (int p = 0; p < 8; ++p)
          ap.u[p] = *(const unsigned*)&p_s[16 * tr + col][32 * kb2 + a_kpair(p, g)];
        o_acc[i] = __builtin_amdgcn_wmma_f32_16x16x32_bf16(false, ap.v, false, bv.v,
                                                           (short)0, o_acc[i], false, false);
      }
    }
    __syncthreads();
  }

  // ---- normalize and store ----
  for (int i = 0; i < ntr; ++i) {
    int tr = trbase + i;
#pragma unroll
    for (int r = 0; r < 8; ++r) {
      int row = 16 * tr + r + 8 * g;
      if (row < NLAT) {
        float v = o_acc[i][r] / l_s[row];
        out[((long)bb * NLAT + row) * INNERq + h * DHq + 16 * tcw + col] = (__bf16)v;
      }
    }
  }
}

// ---------------- host orchestration ----------------
extern "C" void kernel_launch(void* const* d_in, const int* in_sizes, int n_in,
                              void* d_out, int out_size, void* d_ws, size_t ws_size,
                              hipStream_t stream) {
  const float* x        = (const float*)d_in[0];
  const float* pos_emb  = (const float*)d_in[1];
  const float* latents  = (const float*)d_in[2];
  const float* mp_ln_g  = (const float*)d_in[3];
  const float* mp_w     = (const float*)d_in[4];
  const float* mp_b     = (const float*)d_in[5];
  const float* ln_x_w   = (const float*)d_in[6];
  const float* ln_x_b   = (const float*)d_in[7];
  const float* ln_lat_w = (const float*)d_in[8];
  const float* ln_lat_b = (const float*)d_in[9];
  const float* wq       = (const float*)d_in[10];
  const float* wkv      = (const float*)d_in[11];
  const float* wo       = (const float*)d_in[12];
  const float* out_ln_w = (const float*)d_in[13];
  const float* out_ln_b = (const float*)d_in[14];
  const float* ff_ln1_g = (const float*)d_in[15];
  const float* ff_w1    = (const float*)d_in[16];
  const float* ff_ln2_g = (const float*)d_in[17];
  const float* ff_w2    = (const float*)d_in[18];
  (void)in_sizes; (void)n_in; (void)ws_size;

  // workspace carve-up
  char* ws = (char*)d_ws;
  size_t off = 0;
  auto alloc = [&](size_t bytes) -> void* {
    void* p = ws + off;
    off += (bytes + 255) & ~(size_t)255;
    return p;
  };
  const long ROWS = (long)Bq * NLAT;   // 544
  __bf16* xhat     = (__bf16*)alloc((long)Bq * Nn * Dd * 2);
  __bf16* wq_bf    = (__bf16*)alloc((long)DEPTHq * Dd * INNERq * 2);
  __bf16* wkv_bf   = (__bf16*)alloc((long)DEPTHq * Dd * 2 * INNERq * 2);
  __bf16* wo_bf    = (__bf16*)alloc((long)DEPTHq * INNERq * Dd * 2);
  __bf16* ffw1_bf  = (__bf16*)alloc((long)DEPTHq * Dd * FFq * 2);
  __bf16* ffw2_bf  = (__bf16*)alloc((long)DEPTHq * FFq * Dd * 2);
  __bf16* wkvf     = (__bf16*)alloc((long)Dd * 2 * INNERq * 2);
  float*  kvbias   = (float*)alloc(2 * INNERq * 4);
  float*  mp       = (float*)alloc((long)Bq * Dd * 4);
  float*  lnmp     = (float*)alloc((long)Bq * Dd * 4);
  float*  mplat    = (float*)alloc((long)Bq * 4 * Dd * 4);
  float*  lat      = (float*)alloc(ROWS * Dd * 4);
  __bf16* lnlat_bf = (__bf16*)alloc(ROWS * Dd * 2);
  __bf16* q_bf     = (__bf16*)alloc(ROWS * INNERq * 2);
  __bf16* kvlat_bf = (__bf16*)alloc(ROWS * 2 * INNERq * 2);
  __bf16* kvx_bf   = (__bf16*)alloc((long)Bq * Nn * 2 * INNERq * 2);
  __bf16* aout_bf  = (__bf16*)alloc(ROWS * INNERq * 2);
  float*  proj     = (float*)alloc(ROWS * Dd * 4);
  __bf16* ffin_bf  = (__bf16*)alloc(ROWS * Dd * 2);
  float*  hbuf     = (float*)alloc(ROWS * FFq * 4);
  __bf16* h2_bf    = (__bf16*)alloc(ROWS * FFq * 2);
  float*  ffout    = (float*)alloc(ROWS * Dd * 4);

  auto cast = [&](const float* s, __bf16* d, long n) {
    cast_f32_to_bf16<<<(int)((n + 255) / 256), 256, 0, stream>>>(s, d, n);
  };
  auto gemm = [&](const __bf16* A, const __bf16* B, void* C, const float* bias,
                  int M, int N, int K, float scale, int bf16out) {
    dim3 grid((M + GBM - 1) / GBM, N / GBN);
    gemm_bf16<<<grid, 256, 0, stream>>>(A, B, C, bias, M, N, K, scale, bf16out);
  };

  // --- one-time prep ---
  cast(wq,    wq_bf,   (long)DEPTHq * Dd * INNERq);
  cast(wkv,   wkv_bf,  (long)DEPTHq * Dd * 2 * INNERq);
  cast(wo,    wo_bf,   (long)DEPTHq * INNERq * Dd);
  cast(ff_w1, ffw1_bf, (long)DEPTHq * Dd * FFq);
  cast(ff_w2, ffw2_bf, (long)DEPTHq * FFq * Dd);

  xhat_kernel<<<Bq * Nn, 256, 0, stream>>>(x, pos_emb, xhat);
  mp_mean<<<(Bq * Dd + 255) / 256, 256, 0, stream>>>(x, mp);
  ln_rows<4><<<Bq, 256, 0, stream>>>(mp, mp_ln_g, nullptr, nullptr, lnmp, Dd, 0);
  small_gemm_f32<<<dim3((4 * Dd + 255) / 256, Bq), 256, 0, stream>>>(lnmp, mp_w, mp_b,
                                                                     mplat, Bq, Dd, 4 * Dd);
  lat_init<<<(int)((ROWS * Dd + 255) / 256), 256, 0, stream>>>(mplat, latents, lat);

  // --- layers ---
  for (int l = 0; l < DEPTHq; ++l) {
    const float*  wkv_l  = wkv + (long)l * Dd * 2 * INNERq;
    const __bf16* wq_l   = wq_bf + (long)l * Dd * INNERq;
    const __bf16* wkvb_l = wkv_bf + (long)l * Dd * 2 * INNERq;
    const __bf16* wo_l   = wo_bf + (long)l * INNERq * Dd;
    const __bf16* w1_l   = ffw1_bf + (long)l * Dd * FFq;
    const __bf16* w2_l   = ffw2_bf + (long)l * FFq * Dd;

    // fold LN(x) scale/bias into KV weights
    fold_wkv<<<(int)(((long)Dd * 2 * INNERq + 255) / 256), 256, 0, stream>>>(
        wkv_l, ln_x_w + l * Dd, wkvf);
    kv_bias<<<(2 * INNERq + 255) / 256, 256, 0, stream>>>(wkv_l, ln_x_b + l * Dd, kvbias);

    // LN(lat) -> bf16
    ln_rows<4><<<(int)ROWS, 256, 0, stream>>>(lat, ln_lat_w + l * Dd, ln_lat_b + l * Dd,
                                              lnlat_bf, nullptr, Dd, 0);

    // q (scale folded), kv for latents, kv for tokens
    gemm(lnlat_bf, wq_l,   q_bf,     nullptr, (int)ROWS, INNERq,     Dd, 0.125f, 1);
    gemm(lnlat_bf, wkvb_l, kvlat_bf, nullptr, (int)ROWS, 2 * INNERq, Dd, 1.f, 1);
    gemm(xhat,     wkvf,   kvx_bf,   kvbias,  Bq * Nn,   2 * INNERq, Dd, 1.f, 1);

    // flash attention
    attn_kernel<<<dim3(HEADSq, Bq), 256, 0, stream>>>(q_bf, kvx_bf, kvlat_bf, aout_bf);

    // output proj + LN + residual
    gemm(aout_bf, wo_l, proj, nullptr, (int)ROWS, Dd, INNERq, 1.f, 0);
    postattn_ln_res<<<(int)ROWS, 256, 0, stream>>>(proj, out_ln_w + l * Dd,
                                                   out_ln_b + l * Dd, lat);

    // FF: ln_g -> w1 -> gelu+ln_g -> w2 -> residual
    ln_rows<4><<<(int)ROWS, 256, 0, stream>>>(lat, ff_ln1_g + l * Dd, nullptr,
                                              ffin_bf, nullptr, Dd, 0);
    gemm(ffin_bf, w1_l, hbuf, nullptr, (int)ROWS, FFq, Dd, 1.f, 0);
    ln_rows<16><<<(int)ROWS, 256, 0, stream>>>(hbuf, ff_ln2_g + l * FFq, nullptr,
                                               h2_bf, nullptr, FFq, 1);
    gemm(h2_bf, w2_l, ffout, nullptr, (int)ROWS, Dd, FFq, 1.f, 0);
    add_inplace<<<(int)((ROWS * Dd + 255) / 256), 256, 0, stream>>>(lat, ffout, ROWS * Dd);
  }

  hipMemcpyAsync(d_out, lat, (size_t)out_size * sizeof(float),
                 hipMemcpyDeviceToDevice, stream);
}
